// BatchNormGNNLayer_33492154974255
// MI455X (gfx1250) — matrix operand order repaired and checked by reading, not compiled, not run on hardware
//
#include <hip/hip_runtime.h>
#include <hip/hip_bf16.h>

// ---------------------------------------------------------------------------
// BatchNormGNNLayer for MI455X (gfx1250, wave32, WMMA).
//   agg = segment_sum(x[src], dst)                         (f32 atomics in L2)
//   x1  = agg@W_rel^T + b_rel + x@W_root^T                 (f16 WMMA, f32 acc)
//   x2  = leaky(x1)                -> LDS tile (f16)
//   x3  = x2@W_lin^T + b_lin                               (f16 WMMA, f32 acc)
//   BN(train, biased var) + leaky  -> out                  (f32)
// ---------------------------------------------------------------------------

#define N_NODES 50000
#define N_EDGES 1600000
#define C_IN    128
#define HDIM    128
#define TILE_M  128
#define LDS_STR 136          // halfs per row: 272B -> conflict-free 16B reads
#define NEG_SLOPE 0.01f
#define BN_EPS    1e-5f

typedef _Float16 half_t;
typedef __attribute__((ext_vector_type(16))) _Float16 v16h;
typedef __attribute__((ext_vector_type(8)))  _Float16 v8h;
typedef __attribute__((ext_vector_type(8)))  float    v8f;
typedef __attribute__((ext_vector_type(4)))  float    v4f;

__device__ __forceinline__ float leaky(float v) {
    return v >= 0.0f ? v : NEG_SLOPE * v;
}

// A-fragment (16x32 f16, MxK) from row-major f32 [*, 128].
// lane&15 = row m; (lane>>4)*8 = K sub-offset; chunks at +0 and +16.
__device__ __forceinline__ v16h load_a_f32(const float* __restrict__ base,
                                           int gRowBase, int lane, int kbase) {
    const int m  = lane & 15;
    const int ko = (lane >> 4) << 3;           // 0 or 8
    const int row = gRowBase + m;
    v16h a;
    if (row < N_NODES) {
        const float* p = base + (size_t)row * C_IN + kbase + ko;
        v4f c0 = *(const v4f*)(p);
        v4f c1 = *(const v4f*)(p + 4);
        v4f c2 = *(const v4f*)(p + 16);
        v4f c3 = *(const v4f*)(p + 20);
#pragma unroll
        for (int i = 0; i < 4; ++i) {
            a[i]      = (half_t)c0[i];
            a[4 + i]  = (half_t)c1[i];
            a[8 + i]  = (half_t)c2[i];
            a[12 + i] = (half_t)c3[i];
        }
    } else {
#pragma unroll
        for (int i = 0; i < 16; ++i) a[i] = (half_t)0.0f;
    }
    return a;
}

// A-fragment from the LDS x2 tile (f16, stride LDS_STR halfs).
__device__ __forceinline__ v16h load_a_lds(const half_t* __restrict__ lds,
                                           int rowBase, int lane, int kbase) {
    const int m  = lane & 15;
    const int ko = (lane >> 4) << 3;
    const half_t* p = lds + (size_t)(rowBase + m) * LDS_STR + kbase + ko;
    v8h c0 = *(const v8h*)(p);
    v8h c1 = *(const v8h*)(p + 16);
    v16h a;
#pragma unroll
    for (int i = 0; i < 8; ++i) { a[i] = c0[i]; a[8 + i] = c1[i]; }
    return a;
}

// B-fragment (32x16 f16, KxN): B[k,n] = W[n,k], W row-major [128,128] f16.
// lane&15 = col n; K sub-offset kbase + (lane>>4)*8; chunks at +0 and +16.
__device__ __forceinline__ v16h load_b_w(const half_t* __restrict__ Wh,
                                         int nblock, int lane, int kbase) {
    const int n  = (nblock << 4) + (lane & 15);
    const int kb = kbase + ((lane >> 4) << 3); // +0 or +8
    const half_t* p = Wh + (size_t)n * C_IN + kb;
    v8h b0 = *(const v8h*)(p);
    v8h b1 = *(const v8h*)(p + 16);
    v16h b;
#pragma unroll
    for (int i = 0; i < 8; ++i) { b[i] = b0[i]; b[8 + i] = b1[i]; }
    return b;
}

// ---------------- kernel 1: zero agg + column accumulators --------------------
__global__ __launch_bounds__(256) void k_zero(float* __restrict__ agg,
                                              float* __restrict__ sums) {
    size_t i = (size_t)blockIdx.x * blockDim.x + threadIdx.x;
    const size_t nAgg = (size_t)N_NODES * C_IN;
    if (i < nAgg) agg[i] = 0.0f;
    if (i < 256)  sums[i] = 0.0f;      // colsum[128] ++ colsumsq[128]
}

// ---------------- kernel 2: f32 -> f16 weight conversion ---------------------
__global__ __launch_bounds__(256) void k_cvt_w(const float* __restrict__ Wrel,
                                               const float* __restrict__ Wroot,
                                               const float* __restrict__ Wlin,
                                               half_t* __restrict__ Wh) {
    int i = blockIdx.x * blockDim.x + threadIdx.x;     // 3 * 16384
    if (i >= 3 * HDIM * C_IN) return;
    int which = i / (HDIM * C_IN);
    int j     = i - which * (HDIM * C_IN);
    const float* src = (which == 0) ? Wrel : (which == 1) ? Wroot : Wlin;
    Wh[i] = (half_t)src[j];
}

// ---------------- kernel 3: edge scatter-add (wave per edge) -----------------
__global__ __launch_bounds__(256) void k_scatter(const float* __restrict__ x,
                                                 const long long* __restrict__ ei,
                                                 float* __restrict__ agg) {
    const int wave = threadIdx.x >> 5;
    const int lane = threadIdx.x & 31;
    const long long e = (long long)blockIdx.x * 8 + wave;
    if (e >= N_EDGES) return;
    const int src = (int)ei[e];
    const int dst = (int)ei[(long long)N_EDGES + e];
    // 32 lanes x float4 = 128 contiguous floats gathered per edge
    v4f v = ((const v4f*)(x + (size_t)src * C_IN))[lane];
    float* ad = agg + (size_t)dst * C_IN + lane * 4;
    atomicAdd(ad + 0, v[0]);
    atomicAdd(ad + 1, v[1]);
    atomicAdd(ad + 2, v[2]);
    atomicAdd(ad + 3, v[3]);
}

// ---------------- kernel 4: fused WMMA GEMM1 + leaky + GEMM2 -----------------
__global__ __launch_bounds__(256) void k_gemm_fused(
        const float* __restrict__ x, const float* __restrict__ agg,
        const half_t* __restrict__ Wrel_h, const half_t* __restrict__ Wroot_h,
        const half_t* __restrict__ Wlin_h,
        const float* __restrict__ b_rel, const float* __restrict__ b_lin,
        float* __restrict__ x3,
        float* __restrict__ colsum, float* __restrict__ colsumsq) {
    __shared__ half_t lds[TILE_M * LDS_STR];           // 34,816 B x2 tile

    const int wave  = threadIdx.x >> 5;                // 8 waves: row block
    const int lane  = threadIdx.x & 31;
    const int tile0 = blockIdx.x * TILE_M;
    const int gRow  = tile0 + wave * 16;

    const int col0  = lane & 15;
    const int m_off = (lane >> 4) << 3;                // 0 or 8 (C/D layout)

    // ---- GEMM1: acc = agg@Wrel^T + x@Wroot^T (K=128 each, 32 per step) ----
    v8f acc[8];
#pragma unroll
    for (int nb = 0; nb < 8; ++nb)
#pragma unroll
        for (int r = 0; r < 8; ++r) acc[nb][r] = 0.0f;

#pragma unroll
    for (int kt = 0; kt < 4; ++kt) {
        const int kb = kt * 32;
        v16h aA = load_a_f32(agg, gRow, lane, kb);
        v16h aX = load_a_f32(x,   gRow, lane, kb);
#pragma unroll
        for (int nb = 0; nb < 8; ++nb) {
            v16h bR = load_b_w(Wrel_h,  nb, lane, kb);
            acc[nb] = __builtin_amdgcn_wmma_f32_16x16x32_f16(
                          false, aA, false, bR, (short)0, acc[nb], false, false);
            v16h bT = load_b_w(Wroot_h, nb, lane, kb);
            acc[nb] = __builtin_amdgcn_wmma_f32_16x16x32_f16(
                          false, aX, false, bT, (short)0, acc[nb], false, false);
        }
    }

    // ---- bias + leaky, park x2 tile in LDS as f16 ----
#pragma unroll
    for (int nb = 0; nb < 8; ++nb) {
        const int col = (nb << 4) + col0;
        const float bias = b_rel[col];
#pragma unroll
        for (int r = 0; r < 8; ++r) {
            float v = leaky(acc[nb][r] + bias);
            lds[(size_t)(wave * 16 + m_off + r) * LDS_STR + col] = (half_t)v;
        }
    }
    __syncthreads();

    // ---- GEMM2: acc2 = x2 @ Wlin^T ----
    v8f acc2[8];
#pragma unroll
    for (int nb = 0; nb < 8; ++nb)
#pragma unroll
        for (int r = 0; r < 8; ++r) acc2[nb][r] = 0.0f;

#pragma unroll
    for (int kt = 0; kt < 4; ++kt) {
        const int kb = kt * 32;
        v16h a2 = load_a_lds(lds, wave * 16, lane, kb);
#pragma unroll
        for (int nb = 0; nb < 8; ++nb) {
            v16h bL = load_b_w(Wlin_h, nb, lane, kb);
            acc2[nb] = __builtin_amdgcn_wmma_f32_16x16x32_f16(
                           false, a2, false, bL, (short)0, acc2[nb], false, false);
        }
    }

    // ---- bias, store x3, accumulate per-column sum / sum^2 ----
#pragma unroll
    for (int nb = 0; nb < 8; ++nb) {
        const int col = (nb << 4) + col0;
        const float bias = b_lin[col];
        float ps = 0.0f, pss = 0.0f;
#pragma unroll
        for (int r = 0; r < 8; ++r) {
            const int row = gRow + m_off + r;
            if (row < N_NODES) {
                float v = acc2[nb][r] + bias;
                x3[(size_t)row * HDIM + col] = v;
                ps  += v;
                pss += v * v;
            }
        }
        atomicAdd(&colsum[col],   ps);
        atomicAdd(&colsumsq[col], pss);
    }
}

// ---------------- kernel 5: BatchNorm finalize + leaky -----------------------
__global__ __launch_bounds__(256) void k_bn(const float* __restrict__ x3,
                                            const float* __restrict__ colsum,
                                            const float* __restrict__ colsumsq,
                                            const float* __restrict__ gamma,
                                            const float* __restrict__ beta,
                                            float* __restrict__ out) {
    size_t idx = (size_t)blockIdx.x * blockDim.x + threadIdx.x;
    if (idx >= (size_t)N_NODES * HDIM) return;
    const int c = (int)(idx & (HDIM - 1));
    const float invN = 1.0f / (float)N_NODES;
    const float mean = colsum[c] * invN;
    const float var  = colsumsq[c] * invN - mean * mean;   // biased batch var
    const float sc   = gamma[c] * rsqrtf(var + BN_EPS);
    out[idx] = leaky((x3[idx] - mean) * sc + beta[c]);
}

// ---------------------------------------------------------------------------
extern "C" void kernel_launch(void* const* d_in, const int* in_sizes, int n_in,
                              void* d_out, int out_size, void* d_ws, size_t ws_size,
                              hipStream_t stream) {
    const float*     x      = (const float*)d_in[0];
    const long long* ei     = (const long long*)d_in[1];   // int64 [2, E]
    // d_in[2] = batch (all zeros) -> unused
    const float*     W_rel  = (const float*)d_in[3];
    const float*     b_rel  = (const float*)d_in[4];
    const float*     W_root = (const float*)d_in[5];
    const float*     W_lin  = (const float*)d_in[6];
    const float*     b_lin  = (const float*)d_in[7];
    const float*     gamma  = (const float*)d_in[8];
    const float*     beta   = (const float*)d_in[9];
    float*           out    = (float*)d_out;

    // workspace layout (f32 elements): agg | x3 | colsum | colsumsq | Wh(f16)
    const size_t nFeat = (size_t)N_NODES * C_IN;           // 6,400,000
    float*  agg      = (float*)d_ws;
    float*  x3       = agg + nFeat;
    float*  colsum   = x3 + nFeat;
    float*  colsumsq = colsum + 128;
    half_t* Wh       = (half_t*)(colsumsq + 128);          // 3 * 16384 halfs
    half_t* Wrel_h   = Wh;
    half_t* Wroot_h  = Wh + HDIM * C_IN;
    half_t* Wlin_h   = Wh + 2 * HDIM * C_IN;

    // 1) zero agg + column accumulators
    {
        const size_t n = nFeat;                 // covers the 256 sums too
        k_zero<<<(unsigned)((n + 255) / 256), 256, 0, stream>>>(agg, colsum);
    }
    // 2) weights f32 -> f16
    k_cvt_w<<<(3 * HDIM * C_IN + 255) / 256, 256, 0, stream>>>(
        W_rel, W_root, W_lin, Wh);
    // 3) edge scatter-add (wave per edge; L2-resident atomics)
    k_scatter<<<N_EDGES / 8, 256, 0, stream>>>(x, ei, agg);
    // 4) fused WMMA GEMM1 + leaky + GEMM2, column stats
    k_gemm_fused<<<(N_NODES + TILE_M - 1) / TILE_M, 256, 0, stream>>>(
        x, agg, Wrel_h, Wroot_h, Wlin_h, b_rel, b_lin, x3, colsum, colsumsq);
    // 5) BN finalize + leaky -> out
    k_bn<<<(unsigned)((nFeat + 255) / 256), 256, 0, stream>>>(
        x3, colsum, colsumsq, gamma, beta, out);
}
